// Attention_57475252355505
// MI455X (gfx1250) — matrix operand order repaired and checked by reading, not compiled, not run on hardware
//
#include <hip/hip_runtime.h>
#include <hip/hip_bf16.h>

// Problem constants (from reference): B=2, Q=K=2048, C=256, H=8, D=32
#define BB 2
#define QQ 2048
#define KKK 2048
#define CC 256
#define HH 8
#define DD 32
#define LDSS 40  // padded half-element row stride for 32-wide LDS tiles (80B, 16B-aligned)

// Attention LDS staging geometry
#define KCH 64   // keys staged per chunk
#define KST 40   // K chunk row stride in halves (32 data + 8 pad) -> conflict-free b128 reads
#define VST 72   // V^T chunk row stride in halves (64 data + 8 pad)

typedef __attribute__((ext_vector_type(16))) _Float16 v16h;
typedef __attribute__((ext_vector_type(8)))  _Float16 v8h;
typedef __attribute__((ext_vector_type(8)))  float    v8f;
typedef __attribute__((ext_vector_type(4)))  float    v4f;

__device__ __forceinline__ v8f wmma32f16(v16h a, v16h b, v8f c) {
  // (neg_a, A, neg_b, B, c_mod, C, reuse_a, reuse_b) -> v_wmma_f32_16x16x32_f16
  return __builtin_amdgcn_wmma_f32_16x16x32_f16(false, a, false, b, (short)0, c, false, false);
}

__device__ __forceinline__ v16h pack16(v8h lo, v8h hi) {
  v16h r;
#pragma unroll
  for (int i = 0; i < 8; ++i) { r[i] = lo[i]; r[i + 8] = hi[i]; }
  return r;
}

__device__ __forceinline__ v8f zero8() {
  v8f z;
#pragma unroll
  for (int i = 0; i < 8; ++i) z[i] = 0.0f;
  return z;
}

// gfx1250 async copy: global -> LDS, 16B per lane, tracked by ASYNCcnt.
__device__ __forceinline__ void async_load_b128(unsigned lds_off, const _Float16* gptr) {
  asm volatile("global_load_async_to_lds_b128 %0, %1, off"
               :: "v"(lds_off), "v"(gptr)
               : "memory");
}
#define WAIT_ASYNC(N) asm volatile("s_wait_asynccnt %0" :: "n"(N) : "memory")

// ---------------------------------------------------------------------------
// Kernel 1/2: dual GEMM  Y1 = X @ W1^T, Y2 = X @ W2^T  (shares X LDS tile).
//  - outA: f16 head layout [B,H,L,D], value * scale          (Q or K path)
//  - gate mode (gateOut != null): Y2 -> sigmoid(Y2+bg) f32 [B*L, C]
//  - kv  mode (gateOut == null):  Y2 -> f16 transposed [B,H,D,L]  (V path)
// ---------------------------------------------------------------------------
__global__ void __launch_bounds__(256)
proj_dual_kernel(const float* __restrict__ X,   // [4096, 256]
                 const float* __restrict__ W1,  // [256, 256]
                 const float* __restrict__ W2,  // [256, 256]
                 _Float16* __restrict__ outA,   // [B,H,L,D] f16
                 _Float16* __restrict__ outBT,  // [B,H,D,L] f16 (kv mode)
                 float* __restrict__ gateOut,   // [B*L, C] f32 (gate mode)
                 const float* __restrict__ bgv, // [C] (gate mode)
                 float scale) {
  __shared__ alignas(16) _Float16 Xs[64 * LDSS];
  __shared__ alignas(16) _Float16 W1s[64 * LDSS];
  __shared__ alignas(16) _Float16 W2s[64 * LDSS];

  const int mblk = blockIdx.x >> 2, nblk = blockIdx.x & 3;
  const int row0 = mblk * 64, col0 = nblk * 64;
  const int wid = threadIdx.x >> 5, lane = threadIdx.x & 31;
  const int wm = wid >> 1, wn = wid & 1;        // 4x2 wave grid -> 16x32 each
  const int hi = lane >> 4, ln = lane & 15;
  const int lr = threadIdx.x >> 2;              // 0..63 tile row for loads
  const int lc = (threadIdx.x & 3) * 8;         // 0,8,16,24

  v8f accA[2] = {zero8(), zero8()};
  v8f accB[2] = {zero8(), zero8()};

  for (int k0 = 0; k0 < CC; k0 += 32) {
    {
      v4f x0 = *(const v4f*)(X + (size_t)(row0 + lr) * CC + k0 + lc);
      v4f x1 = *(const v4f*)(X + (size_t)(row0 + lr) * CC + k0 + lc + 4);
      _Float16* d = &Xs[lr * LDSS + lc];
#pragma unroll
      for (int i = 0; i < 4; ++i) { d[i] = (_Float16)x0[i]; d[i + 4] = (_Float16)x1[i]; }
      v4f a0 = *(const v4f*)(W1 + (size_t)(col0 + lr) * CC + k0 + lc);
      v4f a1 = *(const v4f*)(W1 + (size_t)(col0 + lr) * CC + k0 + lc + 4);
      _Float16* d1 = &W1s[lr * LDSS + lc];
#pragma unroll
      for (int i = 0; i < 4; ++i) { d1[i] = (_Float16)a0[i]; d1[i + 4] = (_Float16)a1[i]; }
      v4f b0 = *(const v4f*)(W2 + (size_t)(col0 + lr) * CC + k0 + lc);
      v4f b1 = *(const v4f*)(W2 + (size_t)(col0 + lr) * CC + k0 + lc + 4);
      _Float16* d2 = &W2s[lr * LDSS + lc];
#pragma unroll
      for (int i = 0; i < 4; ++i) { d2[i] = (_Float16)b0[i]; d2[i + 4] = (_Float16)b1[i]; }
    }
    __syncthreads();

    const _Float16* xr = &Xs[(wm * 16 + ln) * LDSS];
    v16h af = pack16(*(const v8h*)(xr + 16 * hi), *(const v8h*)(xr + 16 * hi + 8));

#pragma unroll
    for (int tt = 0; tt < 2; ++tt) {
      const _Float16* w1r = &W1s[(wn * 32 + tt * 16 + ln) * LDSS + 16 * hi];
      v16h b1 = pack16(*(const v8h*)w1r, *(const v8h*)(w1r + 8));
      accA[tt] = wmma32f16(af, b1, accA[tt]);
      const _Float16* w2r = &W2s[(wn * 32 + tt * 16 + ln) * LDSS + 16 * hi];
      v16h b2 = pack16(*(const v8h*)w2r, *(const v8h*)(w2r + 8));
      accB[tt] = wmma32f16(af, b2, accB[tt]);
    }
    __syncthreads();
  }

#pragma unroll
  for (int tt = 0; tt < 2; ++tt) {
#pragma unroll
    for (int j = 0; j < 8; ++j) {
      const int gm = row0 + wm * 16 + j + 8 * hi;   // global row (b*L + pos)
      const int gn = col0 + wn * 32 + tt * 16 + ln; // global col (h*D + d)
      const int bi = gm >> 11, pos = gm & 2047;     // L == 2048
      const int hh = gn >> 5, dd = gn & 31;         // D == 32
      outA[((size_t)(bi * HH + hh) * QQ + pos) * DD + dd] = (_Float16)(accA[tt][j] * scale);
      if (gateOut != nullptr) {
        float z = accB[tt][j] + bgv[gn];
        gateOut[(size_t)gm * CC + gn] = 1.0f / (1.0f + __expf(-z));
      } else {
        outBT[((size_t)(bi * HH + hh) * DD + dd) * KKK + pos] = (_Float16)accB[tt][j];
      }
    }
  }
}

// ---------------------------------------------------------------------------
// Kernel 3: flash attention with bias + online softmax + gate.
// Block = 8 waves = 8 consecutive 16-query tiles of ONE (b,h).
// K/V chunks (64 keys) are staged in LDS once per block with
// global_load_async_to_lds_b128 (double-buffered, s_wait_asynccnt), so
// K/V L2 traffic drops 8x vs per-wave loading.
// S^T = K_chunk @ Q^T so the softmax axis lives inside a lane (+ shfl_xor 16).
// ---------------------------------------------------------------------------
__global__ void __launch_bounds__(256)
attn_kernel(const _Float16* __restrict__ Qh,  // [B,H,Q,D] f16 (pre-scaled)
            const _Float16* __restrict__ Kh,  // [B,H,K,D] f16
            const _Float16* __restrict__ Vt,  // [B,H,D,K] f16 (transposed)
            const float* __restrict__ bias,   // [B,1,Q,K] f32
            const float* __restrict__ G,      // [B*Q, C] f32 gate
            _Float16* __restrict__ OG) {      // [B*Q, C] f16 gated output
  __shared__ alignas(16) _Float16 Ks[2 * KCH * KST];  // 10240 B
  __shared__ alignas(16) _Float16 Vs[2 * DD * VST];   //  9216 B

  const int wid = threadIdx.x >> 5, lane = threadIdx.x & 31;
  const int bh = blockIdx.x >> 4;             // 16 blocks per (b,h)
  const int qblk = blockIdx.x & 15;
  const int b = bh >> 3, h = bh & 7;
  const int qbase = qblk * 128 + wid * 16;    // this wave's 16-query tile
  const int hi = lane >> 4, ln = lane & 15;

  const _Float16* kb = Kh + (size_t)bh * KKK * DD;
  const _Float16* vb = Vt + (size_t)bh * DD * KKK;

  // --- async staging assignments (per thread: one 16B K seg + one 16B V seg)
  const int t = threadIdx.x;
  const int kr = t >> 2, kseg = t & 3;        // K: row 0..63, 4 segs of 8 halves
  const int vd = t >> 3, vseg = t & 7;        // V: d 0..31, 8 segs of 8 halves
  const unsigned ksBase = (unsigned)(size_t)Ks;
  const unsigned vsBase = (unsigned)(size_t)Vs;
  unsigned ksOff[2], vsOff[2];
#pragma unroll
  for (int u = 0; u < 2; ++u) {
    ksOff[u] = ksBase + (unsigned)((u * KCH * KST + kr * KST + kseg * 8) * 2);
    vsOff[u] = vsBase + (unsigned)((u * DD * VST + vd * VST + vseg * 8) * 2);
  }
  const _Float16* gK = kb + (size_t)kr * DD + kseg * 8;  // + kc*DD per chunk
  const _Float16* gV = vb + (size_t)vd * KKK + vseg * 8; // + kc     per chunk

  // Q^T B-fragment: element e -> k-dim = 16*hi + e, column n = ln
  const _Float16* qrow = Qh + ((size_t)bh * QQ + qbase + ln) * DD + 16 * hi;
  const v16h bq = pack16(*(const v8h*)qrow, *(const v8h*)(qrow + 8));

  const float* brow = bias + ((size_t)b * QQ + qbase + ln) * KKK;
  const float* grow = G + ((size_t)b * QQ + qbase + ln) * CC + h * DD;

  float m_run = -3.0e38f, l_run = 0.0f;
  v8f accO0 = zero8(), accO1 = zero8();

  // prologue: stage chunk 0 into buffer 0
  async_load_b128(ksOff[0], gK);
  async_load_b128(vsOff[0], gV);

  for (int kc = 0; kc < KKK; kc += KCH) {
    const int ibuf = (kc >> 6) & 1;
    if (kc + KCH < KKK) {  // prefetch next chunk into the other buffer
      async_load_b128(ksOff[ibuf ^ 1], gK + (size_t)(kc + KCH) * DD);
      async_load_b128(vsOff[ibuf ^ 1], gV + (kc + KCH));
      __builtin_prefetch((const void*)(brow + kc + KCH), 0, 1);
      WAIT_ASYNC(2);       // wait for THIS chunk (2 newer remain outstanding)
    } else {
      WAIT_ASYNC(0);
    }
    __syncthreads();       // all waves' async data now visible in LDS

    const _Float16* ksb = &Ks[ibuf * KCH * KST];
    const _Float16* vsb = &Vs[ibuf * DD * VST];

#pragma unroll
    for (int s = 0; s < 2; ++s) {            // two 32-key subchunks
      const int kk = kc + s * 32;
      // K A-fragments from LDS: tile tt rows k = kk + tt*16 + m
      const _Float16* r0 = ksb + (s * 32 + ln) * KST;
      v16h ak0 = pack16(*(const v8h*)(r0 + 16 * hi), *(const v8h*)(r0 + 16 * hi + 8));
      const _Float16* r1 = ksb + (s * 32 + 16 + ln) * KST;
      v16h ak1 = pack16(*(const v8h*)(r1 + 16 * hi), *(const v8h*)(r1 + 16 * hi + 8));

      v8f s0 = wmma32f16(ak0, bq, zero8());  // S^T rows k: kk+{0..15}
      v8f s1 = wmma32f16(ak1, bq, zero8());  // S^T rows k: kk+{16..31}

      // bias: element (m=j+8*hi -> k, n=ln -> q); contiguous in j -> float4s
      v4f b00 = *(const v4f*)(brow + kk + 8 * hi);
      v4f b01 = *(const v4f*)(brow + kk + 8 * hi + 4);
      v4f b10 = *(const v4f*)(brow + kk + 16 + 8 * hi);
      v4f b11 = *(const v4f*)(brow + kk + 16 + 8 * hi + 4);

      float mx = -3.0e38f;
#pragma unroll
      for (int j = 0; j < 4; ++j) {
        s0[j] += b00[j]; s0[j + 4] += b01[j];
        s1[j] += b10[j]; s1[j + 4] += b11[j];
      }
#pragma unroll
      for (int j = 0; j < 8; ++j) mx = fmaxf(mx, fmaxf(s0[j], s1[j]));
      mx = fmaxf(mx, __shfl_xor(mx, 16, 32));  // column max over all 32 keys

      const float m_new = fmaxf(m_run, mx);
      const float alpha = __expf(m_run - m_new);
      float p0[8], p1[8], rs = 0.0f;
#pragma unroll
      for (int j = 0; j < 8; ++j) {
        p0[j] = __expf(s0[j] - m_new);
        p1[j] = __expf(s1[j] - m_new);
        rs += p0[j] + p1[j];
      }
      rs += __shfl_xor(rs, 16, 32);
      l_run = l_run * alpha + rs;
      m_run = m_new;
#pragma unroll
      for (int j = 0; j < 8; ++j) { accO0[j] *= alpha; accO1[j] *= alpha; }

      // Convert P^T C-fragments -> B-fragment (32k x 16q) via lane-16 swap
      float sh0[8], sh1[8];
#pragma unroll
      for (int j = 0; j < 8; ++j) {
        sh0[j] = __shfl_xor(p0[j], 16, 32);
        sh1[j] = __shfl_xor(p1[j], 16, 32);
      }
      v16h pf;
#pragma unroll
      for (int e = 0; e < 8; ++e) {
        pf[e]     = (_Float16)(hi ? sh1[e] : p0[e]);  // k = 16*hi + e
        pf[e + 8] = (_Float16)(hi ? p1[e] : sh0[e]);  // k = 16*hi + 8 + e
      }

      // V^T A-fragments from LDS (d as M, k as K)
      const _Float16* v0 = vsb + ln * VST + s * 32;
      const _Float16* v1 = vsb + (16 + ln) * VST + s * 32;
      v16h av0 = pack16(*(const v8h*)(v0 + 16 * hi), *(const v8h*)(v0 + 16 * hi + 8));
      v16h av1 = pack16(*(const v8h*)(v1 + 16 * hi), *(const v8h*)(v1 + 16 * hi + 8));
      accO0 = wmma32f16(av0, pf, accO0);  // O^T rows d 0..15
      accO1 = wmma32f16(av1, pf, accO1);  // O^T rows d 16..31
    }
    __syncthreads();  // protect buffers before next prefetch overwrites
  }

  // normalize, gate, store f16 [B*Q, C]
  const float inv = 1.0f / l_run;
  v4f g00 = *(const v4f*)(grow + 8 * hi);
  v4f g01 = *(const v4f*)(grow + 8 * hi + 4);
  v4f g10 = *(const v4f*)(grow + 16 + 8 * hi);
  v4f g11 = *(const v4f*)(grow + 16 + 8 * hi + 4);
  v8h o0, o1;
#pragma unroll
  for (int j = 0; j < 4; ++j) {
    o0[j]     = (_Float16)(accO0[j] * inv * g00[j]);
    o0[j + 4] = (_Float16)(accO0[j + 4] * inv * g01[j]);
    o1[j]     = (_Float16)(accO1[j] * inv * g10[j]);
    o1[j + 4] = (_Float16)(accO1[j + 4] * inv * g11[j]);
  }
  _Float16* orow = OG + ((size_t)b * QQ + qbase + ln) * CC + h * DD;
  *(v8h*)(orow + 8 * hi) = o0;
  *(v8h*)(orow + 16 + 8 * hi) = o1;
}

// ---------------------------------------------------------------------------
// Kernel 4: output projection  Y = OG @ Wo^T + bo  (f32 out -> d_out)
// ---------------------------------------------------------------------------
__global__ void __launch_bounds__(256)
out_proj_kernel(const _Float16* __restrict__ Xh,  // [4096, 256] f16
                const float* __restrict__ Wo,     // [256, 256]
                const float* __restrict__ bo,     // [256]
                float* __restrict__ Y) {          // [4096, 256] f32
  __shared__ alignas(16) _Float16 Xs[64 * LDSS];
  __shared__ alignas(16) _Float16 Ws[64 * LDSS];
  const int mblk = blockIdx.x >> 2, nblk = blockIdx.x & 3;
  const int row0 = mblk * 64, col0 = nblk * 64;
  const int wid = threadIdx.x >> 5, lane = threadIdx.x & 31;
  const int wm = wid >> 1, wn = wid & 1;
  const int hi = lane >> 4, ln = lane & 15;
  const int lr = threadIdx.x >> 2, lc = (threadIdx.x & 3) * 8;

  v8f acc0 = zero8(), acc1 = zero8();
  for (int k0 = 0; k0 < CC; k0 += 32) {
    *(v8h*)&Xs[lr * LDSS + lc] = *(const v8h*)(Xh + (size_t)(row0 + lr) * CC + k0 + lc);
    v4f w0 = *(const v4f*)(Wo + (size_t)(col0 + lr) * CC + k0 + lc);
    v4f w1 = *(const v4f*)(Wo + (size_t)(col0 + lr) * CC + k0 + lc + 4);
    _Float16* wd = &Ws[lr * LDSS + lc];
#pragma unroll
    for (int i = 0; i < 4; ++i) { wd[i] = (_Float16)w0[i]; wd[i + 4] = (_Float16)w1[i]; }
    __syncthreads();
    const _Float16* xr = &Xs[(wm * 16 + ln) * LDSS];
    v16h af = pack16(*(const v8h*)(xr + 16 * hi), *(const v8h*)(xr + 16 * hi + 8));
    const _Float16* w0r = &Ws[(wn * 32 + ln) * LDSS + 16 * hi];
    acc0 = wmma32f16(af, pack16(*(const v8h*)w0r, *(const v8h*)(w0r + 8)), acc0);
    const _Float16* w1r = &Ws[(wn * 32 + 16 + ln) * LDSS + 16 * hi];
    acc1 = wmma32f16(af, pack16(*(const v8h*)w1r, *(const v8h*)(w1r + 8)), acc1);
    __syncthreads();
  }
#pragma unroll
  for (int j = 0; j < 8; ++j) {
    const int gm = row0 + wm * 16 + j + 8 * hi;
    const int gn = col0 + wn * 32 + ln;
    Y[(size_t)gm * CC + gn] = acc0[j] + bo[gn];
    Y[(size_t)gm * CC + gn + 16] = acc1[j] + bo[gn + 16];
  }
}

// ---------------------------------------------------------------------------
// Host launcher. Workspace layout (needs 12 MB):
//   [0,2MB)   Qh  f16 [B,H,Q,D]   (pre-scaled by 1/sqrt(D))
//   [2,4MB)   Kh  f16 [B,H,K,D]
//   [4,6MB)   Vt  f16 [B,H,D,K]   (transposed for V^T A-fragments)
//   [6,10MB)  G   f32 [B*Q, C]    (sigmoid gate)
//   [10,12MB) OG  f16 [B*Q, C]    (gated attention output)
// ---------------------------------------------------------------------------
extern "C" void kernel_launch(void* const* d_in, const int* in_sizes, int n_in,
                              void* d_out, int out_size, void* d_ws, size_t ws_size,
                              hipStream_t stream) {
  (void)in_sizes; (void)n_in; (void)out_size; (void)ws_size;
  const float* q_x  = (const float*)d_in[0];
  const float* kv_x = (const float*)d_in[1];
  const float* bias = (const float*)d_in[2];
  const float* Wq   = (const float*)d_in[3];
  const float* Wk   = (const float*)d_in[4];
  const float* Wv   = (const float*)d_in[5];
  const float* Wo   = (const float*)d_in[6];
  const float* bo   = (const float*)d_in[7];
  const float* Wg   = (const float*)d_in[8];
  const float* bg   = (const float*)d_in[9];
  float* out = (float*)d_out;

  char* ws = (char*)d_ws;
  _Float16* Qh = (_Float16*)(ws);
  _Float16* Kh = (_Float16*)(ws + (2u << 20));
  _Float16* Vt = (_Float16*)(ws + (4u << 20));
  float*    G  = (float*)   (ws + (6u << 20));
  _Float16* OG = (_Float16*)(ws + (10u << 20));

  const float scale = 0.17677669529663688f;  // 1/sqrt(32)

  dim3 blk(256);
  // Q + gate from q_x
  proj_dual_kernel<<<256, blk, 0, stream>>>(q_x, Wq, Wg, Qh, nullptr, G, bg, scale);
  // K + V (transposed) from kv_x
  proj_dual_kernel<<<256, blk, 0, stream>>>(kv_x, Wk, Wv, Kh, Vt, nullptr, nullptr, 1.0f);
  // flash attention + gating (async LDS staging, double buffered)
  attn_kernel<<<256, blk, 0, stream>>>(Qh, Kh, Vt, bias, G, OG);
  // output projection
  out_proj_kernel<<<256, blk, 0, stream>>>(OG, Wo, bo, out);
}
